// EdgeUpdate_63668595196291
// MI455X (gfx1250) — matrix operand order repaired and checked by reading, not compiled, not run on hardware
//
#include <hip/hip_runtime.h>

typedef _Float16 v16h __attribute__((ext_vector_type(16)));
typedef _Float16 v8h  __attribute__((ext_vector_type(8)));
typedef float    v8f  __attribute__((ext_vector_type(8)));
typedef unsigned int v4u __attribute__((ext_vector_type(4)));

#define WAVES_PER_BLOCK 8
#define BLOCK_THREADS (WAVES_PER_BLOCK * 32)
#define HSTRIDE 72   // halves per LDS row (144B = 16B aligned, conflict-free b128 reads)

// packed f16 relu without the compiler's canonicalize pre-pass:
// exactly one v_pk_max_num_f16 per 32-bit register
__device__ __forceinline__ v8h relu8h(v8h x) {
    v4u u = __builtin_bit_cast(v4u, x);
#pragma unroll
    for (int i = 0; i < 4; ++i) {
        unsigned int r;
        asm("v_pk_max_num_f16 %0, %1, 0" : "=v"(r) : "v"(u[i]));
        u[i] = r;
    }
    return __builtin_bit_cast(v8h, u);
}

__global__ __launch_bounds__(BLOCK_THREADS) void edge_mlp_wmma(
    const float* __restrict__ src, const float* __restrict__ dst,
    const float* __restrict__ ea,
    const float* __restrict__ W1, const float* __restrict__ b1,
    const float* __restrict__ W2, const float* __restrict__ b2,
    const float* __restrict__ W3, const float* __restrict__ b3,
    float* __restrict__ out, int E, int numTiles, int totalWaves)
{
    __shared__ __align__(16) _Float16 hbuf[WAVES_PER_BLOCK][16 * HSTRIDE];
    __shared__ float cbuf[WAVES_PER_BLOCK][16 * 2];

    const int lane = threadIdx.x & 31;
    const int wave = threadIdx.x >> 5;
    const int half = lane >> 4;     // which 16-lane half of the wave
    const int l16  = lane & 15;
    const int waveGid = blockIdx.x * WAVES_PER_BLOCK + wave;
    const bool h0 = (half == 0);

    // ---------------- persistent weight fragments (B-layout: lane=col n, half=K-range) ----
    // Layer 1 (4->32): zero-padded to K=32; only K rows 0..3 (lanes 0-15, i<4) are nonzero.
    v16h B1a = {}, B1b = {};
    if (h0) {
        for (int i = 0; i < 4; ++i) {
            B1a[i] = (_Float16)W1[i * 32 + l16];
            B1b[i] = (_Float16)W1[i * 32 + 16 + l16];
        }
    }
    // Layer 2 (32->32): full K=32 fragments, two N-halves.
    v16h B2a, B2b;
    for (int i = 0; i < 16; ++i) {
        int k = (i < 8) ? (half * 8 + i) : (16 + half * 8 + (i - 8));
        B2a[i] = (_Float16)W2[k * 32 + l16];
        B2b[i] = (_Float16)W2[k * 32 + 16 + l16];
    }
    // Layer 3 (32->2): N zero-padded to 16 (only cols 0,1 valid).
    v16h B3 = {};
    if (l16 < 2) {
        for (int i = 0; i < 16; ++i) {
            int k = (i < 8) ? (half * 8 + i) : (16 + half * 8 + (i - 8));
            B3[i] = (_Float16)W3[k * 2 + l16];
        }
    }

    // persistent bias accumulator fragments
    v8h Cb1a, Cb1b, Cb2a, Cb2b;
    v8f Cb3;
    {
        const _Float16 h1a = (_Float16)b1[l16], h1b = (_Float16)b1[16 + l16];
        const _Float16 h2a = (_Float16)b2[l16], h2b = (_Float16)b2[16 + l16];
        for (int j = 0; j < 8; ++j) {
            Cb1a[j] = h1a; Cb1b[j] = h1b;
            Cb2a[j] = h2a; Cb2b[j] = h2b;
        }
        const float b3v = (l16 < 2) ? b3[l16] : 0.0f;
        for (int r = 0; r < 8; ++r) Cb3[r] = b3v;
    }

    _Float16* hrow = &hbuf[wave][0];
    float*    crow = &cbuf[wave][0];
    const int kb = half * 8;       // A-layout K-base for this lane half
    const int mb = half * 8;       // f16 D-layout row base for this lane half

    const _Float16 hz = (_Float16)0.0f;
    v16h A1 = {};                  // elements 4..15 stay zero for the whole kernel

    for (int t = waveGid; t < numTiles; t += totalWaves) {
        int e = t * 16 + l16;
        const bool valid = (e < E);
        if (!valid) e = E - 1;

        // prefetch next tile for this wave
        int ep = e + totalWaves * 16;
        if (ep < E) {
            __builtin_prefetch(&src[ep], 0, 0);
            __builtin_prefetch(&dst[ep], 0, 0);
            __builtin_prefetch(&ea[ep * 2], 0, 0);
        }

        // ----- per-edge pointwise features (both halves compute; loads coalesce) -----
        const float s    = src[e];
        const float d    = dst[e];
        const float e0   = ea[e * 2 + 0];
        const float e1   = ea[e * 2 + 1];
        const float diff = s - d;
        const float n2   = e0 * e0 + e1 * e1;
        const float nrm  = __builtin_amdgcn_sqrtf(n2);       // raw v_sqrt_f32
        const float inv  = diff * __builtin_amdgcn_rcpf(n2); // raw v_rcp_f32 + mul
        const float inp0 = inv * e0;
        const float inp1 = inv * e1;

        // ----- layer 1: A = zero-padded feats, branchless per-lane select ------------
        A1[0] = h0 ? (_Float16)diff : hz;
        A1[1] = h0 ? (_Float16)e0   : hz;
        A1[2] = h0 ? (_Float16)e1   : hz;
        A1[3] = h0 ? (_Float16)nrm  : hz;

        // f16-accumulate WMMA: D arrives already in f16, bias preloaded
        v8h H1a = __builtin_amdgcn_wmma_f16_16x16x32_f16(false, A1, false, B1a, (short)0, Cb1a, false, false);
        v8h H1b = __builtin_amdgcn_wmma_f16_16x16x32_f16(false, A1, false, B1b, (short)0, Cb1b, false, false);
        H1a = relu8h(H1a);
        H1b = relu8h(H1b);

        // scatter to LDS row-major [m][k]; f16 D-layout: element j = row (mb+j), col l16
        for (int j = 0; j < 8; ++j) {
            hrow[(mb + j) * HSTRIDE + l16]      = H1a[j];
            hrow[(mb + j) * HSTRIDE + 16 + l16] = H1b[j];
        }
        asm volatile("s_wait_dscnt 0" ::: "memory");
        // gather A-layout fragment: lane m reads K {kb..kb+7} and {16+kb..16+kb+7}
        v8h lo = *(const v8h*)&hrow[l16 * HSTRIDE + kb];
        v8h hi = *(const v8h*)&hrow[l16 * HSTRIDE + 16 + kb];
        v16h A2 = __builtin_shufflevector(lo, hi, 0,1,2,3,4,5,6,7,8,9,10,11,12,13,14,15);

        // ----- layer 2: the main 16x32 @ 32x32 GEMM, f16 accumulate + packed relu -----
        v8h H2a = __builtin_amdgcn_wmma_f16_16x16x32_f16(false, A2, false, B2a, (short)0, Cb2a, false, false);
        v8h H2b = __builtin_amdgcn_wmma_f16_16x16x32_f16(false, A2, false, B2b, (short)0, Cb2b, false, false);
        H2a = relu8h(H2a);
        H2b = relu8h(H2b);

        for (int j = 0; j < 8; ++j) {
            hrow[(mb + j) * HSTRIDE + l16]      = H2a[j];
            hrow[(mb + j) * HSTRIDE + 16 + l16] = H2b[j];
        }
        asm volatile("s_wait_dscnt 0" ::: "memory");
        v8h lo3 = *(const v8h*)&hrow[l16 * HSTRIDE + kb];
        v8h hi3 = *(const v8h*)&hrow[l16 * HSTRIDE + 16 + kb];
        v16h A3 = __builtin_shufflevector(lo3, hi3, 0,1,2,3,4,5,6,7,8,9,10,11,12,13,14,15);

        // ----- layer 3: 16x32 @ 32x2 (N zero-padded to 16), f32 accumulate for coeffs --
        v8f C3 = __builtin_amdgcn_wmma_f32_16x16x32_f16(false, A3, false, B3, (short)0, Cb3, false, false);

        // redistribute coeffs through LDS: lanes with col<2 hold valid output columns
        if (l16 < 2) {
            for (int r = 0; r < 8; ++r)
                crow[(r + half * 8) * 2 + l16] = C3[r];
        }
        asm volatile("s_wait_dscnt 0" ::: "memory");
        const float c0 = crow[l16 * 2 + 0];
        const float c1 = crow[l16 * 2 + 1];

        if (half == 0 && valid) {
            float2 o;
            o.x = inp0 * c0;
            o.y = inp1 * c1;
            *(float2*)(&out[e * 2]) = o;   // coalesced 8B store per edge
        }
    }
}

extern "C" void kernel_launch(void* const* d_in, const int* in_sizes, int n_in,
                              void* d_out, int out_size, void* d_ws, size_t ws_size,
                              hipStream_t stream) {
    const float* src  = (const float*)d_in[0];
    const float* dst  = (const float*)d_in[1];
    const float* ea   = (const float*)d_in[2];
    // d_in[3] = u (unused by the math for out), d_in[4] = batch (unused)
    const float* W1   = (const float*)d_in[5];
    const float* b1   = (const float*)d_in[6];
    const float* W2   = (const float*)d_in[7];
    const float* b2   = (const float*)d_in[8];
    const float* W3   = (const float*)d_in[9];
    const float* b3   = (const float*)d_in[10];
    float* out = (float*)d_out;

    const int E = in_sizes[0];
    const int numTiles = (E + 15) / 16;
    int blocks = 2048;
    int maxBlocks = (numTiles + WAVES_PER_BLOCK - 1) / WAVES_PER_BLOCK;
    if (blocks > maxBlocks) blocks = maxBlocks;
    const int totalWaves = blocks * WAVES_PER_BLOCK;

    edge_mlp_wmma<<<blocks, BLOCK_THREADS, 0, stream>>>(
        src, dst, ea, W1, b1, W2, b2, W3, b3, out, E, numTiles, totalWaves);
}
